// MultiHeadAttention_17643725652075
// MI455X (gfx1250) — matrix operand order repaired and checked by reading, not compiled, not run on hardware
//
#include <hip/hip_runtime.h>
#include <hip/hip_bf16.h>

typedef _Float16 half_t;
typedef __attribute__((ext_vector_type(16))) _Float16 v16h;
typedef __attribute__((ext_vector_type(8)))  _Float16 v8h;
typedef __attribute__((ext_vector_type(8)))  float    v8f;

#define D_MODEL  1024
#define N_HEADS  16
#define HEAD_DIM 64
#define T_SEQ    2048
#define BATCH    4
#define NROWS    (BATCH * T_SEQ)   // 8192 token rows

// ---------------------------------------------------------------------------
// CDNA5 wave32 WMMA fragment layouts (CDNA5 ISA §7.12.2):
// A 16x32 f16: lane m=lane&15; code's own order: one contiguous 16-half chunk at k0+16*hi
// B 32x16 f16: lane n=lane&15; code's own order: one contiguous 16-half chunk at k0+16*hi, as A.
// C/D 16x16 f32: element i -> (row = i + 8*hi, col = lane&15)
// ---------------------------------------------------------------------------
static __device__ __forceinline__ v8f wmma_f16(v16h a, v16h b, v8f c) {
    return __builtin_amdgcn_wmma_f32_16x16x32_f16(false, a, false, b,
                                                  (short)0, c, false, false);
}

static __device__ __forceinline__ v16h pack16(v8h lo, v8h hh) {
    union { v16h v; v8h h[2]; } u;
    u.h[0] = lo; u.h[1] = hh;
    return u.v;
}

// ---------------------------------------------------------------------------
// fp32 -> f16 conversion
// ---------------------------------------------------------------------------
__global__ __launch_bounds__(256)
void f32_to_f16_kernel(const float* __restrict__ in, half_t* __restrict__ out, int n) {
    int i = blockIdx.x * blockDim.x + threadIdx.x;
    int stride = gridDim.x * blockDim.x;
    for (; i < n; i += stride) out[i] = (half_t)in[i];
}

// ---------------------------------------------------------------------------
// GEMM: Y[M,N] = A[M,K] * W[N,K]^T  (f16, K contiguous in both)
// Block = 256 threads = 8 waves (4M x 2N); wave tile = 32x64 (8 accumulators).
// MODE 0: Y f16 row-major (Q projection; K==Q per reference bug)
// MODE 1: Y f16 V-transposed [ (b*H+h)*64+d ][ T ] (V projection)
// MODE 2: Y f32 row-major (final projection -> d_out)
// ---------------------------------------------------------------------------
template<int MODE>
__global__ __launch_bounds__(256)
void gemm_wmma_kernel(const half_t* __restrict__ A, const half_t* __restrict__ W,
                      void* __restrict__ Y, int N, int K) {
    const int wid  = threadIdx.x >> 5;
    const int lane = threadIdx.x & 31;
    const int hi = lane >> 4, n = lane & 15;

    const int mrow0 = blockIdx.x * 128 + (wid >> 1) * 32;
    const int ncol0 = blockIdx.y * 128 + (wid & 1) * 64;

    // Per-lane 32-bit element offsets; only the uniform base advances with k.
    const int aoff0 = (mrow0 + n) * K + 16 * hi;          // A tile mt=0 (m == n bits)
    const int aoff1 = aoff0 + 16 * K;                    // A tile mt=1
    int boff[4];
#pragma unroll
    for (int t = 0; t < 4; ++t)
        boff[t] = (ncol0 + t * 16 + n) * K + 16 * hi;

    v8f acc[2][4];
#pragma unroll
    for (int mt = 0; mt < 2; ++mt)
#pragma unroll
        for (int nt = 0; nt < 4; ++nt)
            acc[mt][nt] = (v8f)0.0f;

#pragma unroll 2
    for (int k0 = 0; k0 < K; k0 += 32) {
        const half_t* Ak = A + k0;   // uniform (SGPR) base
        const half_t* Wk = W + k0;

        v16h a0 = pack16(*(const v8h*)(Ak + aoff0), *(const v8h*)(Ak + aoff0 + 8));
        v16h a1 = pack16(*(const v8h*)(Ak + aoff1), *(const v8h*)(Ak + aoff1 + 8));
        v16h b0 = *(const v16h*)(Wk + boff[0]);
        v16h b1 = *(const v16h*)(Wk + boff[1]);
        v16h b2 = *(const v16h*)(Wk + boff[2]);
        v16h b3 = *(const v16h*)(Wk + boff[3]);

        acc[0][0] = wmma_f16(a0, b0, acc[0][0]);
        acc[0][1] = wmma_f16(a0, b1, acc[0][1]);
        acc[0][2] = wmma_f16(a0, b2, acc[0][2]);
        acc[0][3] = wmma_f16(a0, b3, acc[0][3]);
        acc[1][0] = wmma_f16(a1, b0, acc[1][0]);
        acc[1][1] = wmma_f16(a1, b1, acc[1][1]);
        acc[1][2] = wmma_f16(a1, b2, acc[1][2]);
        acc[1][3] = wmma_f16(a1, b3, acc[1][3]);
    }

    // Epilogue: one pointer per 16x16 tile, 8 constant-stride stores.
#pragma unroll
    for (int mt = 0; mt < 2; ++mt) {
#pragma unroll
        for (int nt = 0; nt < 4; ++nt) {
            const int row0 = mrow0 + mt * 16 + 8 * hi;    // + i, i = 0..7
            const int col  = ncol0 + nt * 16 + n;
            if (MODE == 0) {
                half_t* p = (half_t*)Y + (size_t)row0 * N + col;
#pragma unroll
                for (int i = 0; i < 8; ++i)
                    p[(size_t)i * N] = (half_t)acc[mt][nt][i];
            } else if (MODE == 1) {
                // rows of a tile never cross a batch boundary (16 | 2048)
                const int b = row0 >> 11, t0 = row0 & (T_SEQ - 1);
                const int h = col >> 6,  d  = col & (HEAD_DIM - 1);
                half_t* p = (half_t*)Y +
                    ((size_t)((b * N_HEADS + h) * HEAD_DIM + d)) * T_SEQ + t0;
#pragma unroll
                for (int i = 0; i < 8; ++i)
                    p[i] = (half_t)acc[mt][nt][i];       // t = t0 + i
            } else {
                float* p = (float*)Y + (size_t)row0 * N + col;
#pragma unroll
                for (int i = 0; i < 8; ++i)
                    p[(size_t)i * N] = acc[mt][nt][i];
            }
        }
    }
}

// ---------------------------------------------------------------------------
// Causal flash attention. One wave = one (b, h, 16-row q-tile); waves are
// fully independent (no block barrier). Scores via WMMA, online softmax with
// width-16 xor-shuffle reductions (matches C-fragment lane-half row split),
// P staged through LDS (C-layout -> A-fragment layout), PV via WMMA against
// transposed V (key axis contiguous).
// ---------------------------------------------------------------------------
__global__ __launch_bounds__(256)
void attn_wmma_kernel(const half_t* __restrict__ Q, const half_t* __restrict__ Vt,
                      half_t* __restrict__ Out) {
    __shared__ half_t pbuf[8 * 16 * 32];

    const int wid  = threadIdx.x >> 5;
    const int lane = threadIdx.x & 31;
    const int hi = lane >> 4, n = lane & 15;
    half_t* pw = pbuf + wid * (16 * 32);

    const int task = blockIdx.x * 8 + wid;     // 8192 tasks
    const int b    = task >> 11;
    const int rem  = task & 2047;
    const int h    = rem >> 7;
    const int qt   = rem & 127;
    const int qbase = qt * 16;
    const int rowQ0 = b * T_SEQ + qbase;
    const int dcol0 = h * HEAD_DIM;

    // Q tile fragments (reused across the whole key loop)
    const half_t* qp = Q + (size_t)(rowQ0 + n) * D_MODEL + dcol0 + 16 * hi;
    v16h aq0 = pack16(*(const v8h*)qp,        *(const v8h*)(qp + 8));
    v16h aq1 = pack16(*(const v8h*)(qp + 32), *(const v8h*)(qp + 40));

    // Per-lane offsets for K fragments (K == Q) and Vt fragments.
    const int kfo = n * D_MODEL + dcol0 + 16 * hi;   // + {0,32} dchunk, +16*D_MODEL tile
    const int vfo = n * T_SEQ + 16 * hi;             // + t*16*T_SEQ per d-tile
    const half_t* Qb = Q + (size_t)(b * T_SEQ) * D_MODEL;
    const half_t* Vb = Vt + (size_t)((b * N_HEADS + h) * HEAD_DIM) * T_SEQ;

    v8f o[4];
#pragma unroll
    for (int t = 0; t < 4; ++t) o[t] = (v8f)0.0f;
    float m[8], l[8];
#pragma unroll
    for (int i = 0; i < 8; ++i) { m[i] = -__builtin_inff(); l[i] = 0.0f; }

    const int nchunks = (qbase + 16 + 31) >> 5;      // causal: keys <= qbase+15
    for (int ch = 0; ch < nchunks; ++ch) {
        const int kb = ch * 32;
        const half_t* Qk = Qb + (size_t)kb * D_MODEL;   // uniform base
        const half_t* Vk = Vb + kb;

        // ---- S = Q * K^T : two 16-col tiles, accumulate over Dh=64 ----
        v8f c0 = (v8f)0.0f, c1 = (v8f)0.0f;
        c0 = wmma_f16(aq0, *(const v16h*)(Qk + kfo),                   c0);
        c0 = wmma_f16(aq1, *(const v16h*)(Qk + kfo + 32),              c0);
        c1 = wmma_f16(aq0, *(const v16h*)(Qk + kfo + 16 * D_MODEL),    c1);
        c1 = wmma_f16(aq1, *(const v16h*)(Qk + kfo + 16 * D_MODEL + 32), c1);

        // ---- online softmax update ----
#pragma unroll
        for (int i = 0; i < 8; ++i) {
            const int qrow = qbase + i + 8 * hi;
            float s0 = (kb + n      > qrow) ? -__builtin_inff() : c0[i] * 0.125f;
            float s1 = (kb + 16 + n > qrow) ? -__builtin_inff() : c1[i] * 0.125f;

            float mc = fmaxf(s0, s1);
#pragma unroll
            for (int off = 1; off < 16; off <<= 1)
                mc = fmaxf(mc, __shfl_xor(mc, off, 32));
            const float mn = fmaxf(m[i], mc);
            const float al = __expf(m[i] - mn);

            const float p0 = __expf(s0 - mn);
            const float p1 = __expf(s1 - mn);
            float r = p0 + p1;
#pragma unroll
            for (int off = 1; off < 16; off <<= 1)
                r += __shfl_xor(r, off, 32);

            m[i] = mn;
            l[i] = l[i] * al + r;

            pw[(i + 8 * hi) * 32 + n]      = (half_t)p0;
            pw[(i + 8 * hi) * 32 + 16 + n] = (half_t)p1;

#pragma unroll
            for (int t = 0; t < 4; ++t) o[t][i] *= al;
        }

        // ---- read P back as an A fragment (transpose via LDS; DS ops of a
        //      wave are in-order, compiler inserts s_wait_dscnt for RAW) ----
        const half_t* pr = pw + n * 32 + 16 * hi;
        v16h ap = pack16(*(const v8h*)pr, *(const v8h*)(pr + 8));

        // ---- O += P * V ----
#pragma unroll
        for (int t = 0; t < 4; ++t)
            o[t] = wmma_f16(ap, *(const v16h*)(Vk + vfo + t * 16 * T_SEQ), o[t]);
    }

    // ---- normalize and write f16 [B*T, C]: one pointer, stride D_MODEL ----
#pragma unroll
    for (int t = 0; t < 4; ++t) {
        half_t* p = Out + (size_t)(rowQ0 + 8 * hi) * D_MODEL + dcol0 + t * 16 + n;
#pragma unroll
        for (int i = 0; i < 8; ++i)
            p[(size_t)i * D_MODEL] = (half_t)(o[t][i] / l[i]);
    }
}

// ---------------------------------------------------------------------------
extern "C" void kernel_launch(void* const* d_in, const int* in_sizes, int n_in,
                              void* d_out, int out_size, void* d_ws, size_t ws_size,
                              hipStream_t stream) {
    const float* x  = (const float*)d_in[0];
    const float* Wq = (const float*)d_in[1];
    // d_in[2] (W_k) intentionally unused: reference computes K with W_q.
    const float* Wv = (const float*)d_in[3];
    const float* Wo = (const float*)d_in[4];

    const size_t XH = (size_t)NROWS * D_MODEL;     // 8,388,608 halves
    const size_t WH = (size_t)D_MODEL * D_MODEL;   // 1,048,576 halves

    half_t* xh   = (half_t*)d_ws;
    half_t* wqh  = xh   + XH;
    half_t* wvh  = wqh  + WH;
    half_t* woh  = wvh  + WH;
    half_t* Qh   = woh  + WH;   // also serves as K (reference bug: K = x @ W_q^T)
    half_t* Vth  = Qh   + XH;   // V in [b,h,d,t] transposed layout
    half_t* attn = Vth  + XH;   // attention output, f16 [B*T, C]

    f32_to_f16_kernel<<<1024, 256, 0, stream>>>(x,  xh,  (int)XH);
    f32_to_f16_kernel<<<512,  256, 0, stream>>>(Wq, wqh, (int)WH);
    f32_to_f16_kernel<<<512,  256, 0, stream>>>(Wv, wvh, (int)WH);
    f32_to_f16_kernel<<<512,  256, 0, stream>>>(Wo, woh, (int)WH);

    dim3 grid(NROWS / 128, D_MODEL / 128);
    gemm_wmma_kernel<0><<<grid, 256, 0, stream>>>(xh, wqh, Qh,  D_MODEL, D_MODEL);
    gemm_wmma_kernel<1><<<grid, 256, 0, stream>>>(xh, wvh, Vth, D_MODEL, D_MODEL);

    attn_wmma_kernel<<<(BATCH * N_HEADS * (T_SEQ / 16)) / 8, 256, 0, stream>>>(Qh, Vth, attn);

    gemm_wmma_kernel<2><<<grid, 256, 0, stream>>>(attn, woh, d_out, D_MODEL, D_MODEL);
}
